// Attention_18751827214516
// MI455X (gfx1250) — matrix operand order repaired and checked by reading, not compiled, not run on hardware
//
#include <hip/hip_runtime.h>
#include <hip/hip_bf16.h>

typedef __attribute__((ext_vector_type(16))) __bf16 v16bf;
typedef __attribute__((ext_vector_type(8)))  __bf16 v8bf;
typedef __attribute__((ext_vector_type(4)))  __bf16 v4bf;
typedef __attribute__((ext_vector_type(8)))  float  v8f;

#define B_   4
#define S_   1024
#define E_   1024
#define H_   8
#define D_   256
#define NTOK (B_*S_)   // 4096
#define HD   (H_*D_)   // 2048
#define D2   (2*D_)    // 512
#define EPS_ 1e-5f

static __device__ __forceinline__ float  bf2f(__bf16 x){ return (float)x; }
static __device__ __forceinline__ __bf16 f2bf(float x){ return (__bf16)x; }
static __device__ __forceinline__ v16bf cat16(v8bf lo, v8bf hi) {
  return __builtin_shufflevector(lo, hi, 0,1,2,3,4,5,6,7,8,9,10,11,12,13,14,15);
}
static __device__ __forceinline__ float fast_sigmoid(float x) {
  return __builtin_amdgcn_rcpf(1.f + __expf(-x));   // v_rcp_f32, no div expansion
}

// ---------------------------------------------------------------- converts
__global__ void k_f32_to_bf16(const float* __restrict__ in, __bf16* __restrict__ out, int n) {
  int i = blockIdx.x*blockDim.x + threadIdx.x;
  int stride = gridDim.x*blockDim.x;
  for (; i < n; i += stride) out[i] = f2bf(in[i]);
}

// ------------------------------------------------- gate row-sums + logit vec
__global__ __launch_bounds__(256) void k_prep(
    const float* __restrict__ sc_in, const float* __restrict__ ii_in,
    const float* __restrict__ osh_in, const float* __restrict__ ffW2,
    const float* __restrict__ ffb2, const float* __restrict__ attnw,
    const float* __restrict__ attnb,
    float* __restrict__ sc, float* __restrict__ ii, float* __restrict__ osh,
    float* __restrict__ vvec, float* __restrict__ cvec) {
  int h = blockIdx.x, d = threadIdx.x;
  int base = (h*D_ + d)*D_;
  float s0 = 0.f, s1 = 0.f, s2 = 0.f;
  for (int j = 0; j < D_; ++j) { s0 += sc_in[base+j]; s1 += ii_in[base+j]; s2 += osh_in[base+j]; }
  sc[h*D_+d] = s0; ii[h*D_+d] = s1; osh[h*D_+d] = s2;
  float v = 0.f; size_t b2 = (size_t)(h*D_ + d)*E_;
  for (int e = 0; e < E_; ++e) v += ffW2[b2+e]*attnw[e];
  vvec[h*D_+d] = v;
  if (d == 0) {
    float c = 0.f;
    for (int e = 0; e < E_; ++e) c += ffb2[(size_t)h*E_+e]*attnw[e];
    cvec[h] = c + attnb[0];
  }
}

// --------------------------------------------------------- generic WMMA GEMM
// C[M,N](f32|bf16) = A_bf[M,K] @ B_bf[K,N] (+bias[N]) (+exact gelu)
// block = 256 thr (8 waves), block tile 64x64, BK=32, wave tile 16x32.
// B tile stored transposed in LDS so fragments are contiguous 32B reads.
// EPI: 0 = +bias, 1 = +bias then gelu, 2 = none
template<int EPI, typename CT>
__global__ __launch_bounds__(256) void k_gemm(
    const __bf16* __restrict__ A, int lda,
    const __bf16* __restrict__ Bm, int ldb,
    CT* __restrict__ C, int ldc,
    const float* __restrict__ bias, int K) {
  __shared__ __align__(32) __bf16 As[64][32];
  __shared__ __align__(32) __bf16 Bt[64][32];   // [n][k]
  const int tid  = threadIdx.x;
  const int wave = tid >> 5, lane = tid & 31;
  const int wm = wave >> 1, wn = wave & 1;
  const int rbase = blockIdx.x * 64, cbase = blockIdx.y * 64;
  const int r15 = lane & 15, half = lane >> 4;
  const int ar = tid >> 2, ac = (tid & 3) * 8;       // A loader: row/chunk
  const int bn = tid & 63, bk0 = (tid >> 6) * 8;     // B loader: col / k-chunk

  v8f acc0 = {}, acc1 = {};
  for (int kk = 0; kk < K; kk += 32) {
    *(uint4*)&As[ar][ac] = *(const uint4*)&A[(size_t)(rbase+ar)*lda + kk + ac];
    {
      v8bf btmp;                                     // coalesced across lanes (consecutive n)
      #pragma unroll
      for (int j = 0; j < 8; ++j)
        btmp[j] = Bm[(size_t)(kk + bk0 + j)*ldb + cbase + bn];
      *(v8bf*)&Bt[bn][bk0] = btmp;
    }
    if (kk + 32 < K) {  // global_prefetch_b8 the next K tile
      __builtin_prefetch(&A[(size_t)(rbase+ar)*lda + kk + 32 + ac], 0, 1);
      __builtin_prefetch(&Bm[(size_t)(kk+32+bk0)*ldb + cbase + bn], 0, 1);
    }
    __syncthreads();
    // A fragment: two contiguous 8-element runs -> 2x ds_load_b128
    v16bf a;
    {
      const __bf16* ap = &As[wm*16 + r15][half*16];
      a = cat16(*(const v8bf*)ap, *(const v8bf*)(ap + 8));
    }
    // B fragments: contiguous 16 bf16 per lane from transposed tile
    const int c0 = wn*32 + r15;
    v16bf b0 = *(const v16bf*)&Bt[c0][half*16];
    v16bf b1 = *(const v16bf*)&Bt[c0+16][half*16];
    acc0 = __builtin_amdgcn_wmma_f32_16x16x32_bf16(false, a, false, b0, (short)0, acc0, false, false);
    acc1 = __builtin_amdgcn_wmma_f32_16x16x32_bf16(false, a, false, b1, (short)0, acc1, false, false);
    __syncthreads();
  }
  #pragma unroll
  for (int v = 0; v < 8; ++v) {
    int row  = rbase + wm*16 + v + half*8;
    int col0 = cbase + wn*32 + r15;
    float x0 = acc0[v], x1 = acc1[v];
    if (EPI != 2) { x0 += bias[col0]; x1 += bias[col0+16]; }
    if (EPI == 1) {
      x0 = 0.5f*x0*(1.0f + erff(x0*0.70710678f));
      x1 = 0.5f*x1*(1.0f + erff(x1*0.70710678f));
    }
    C[(size_t)row*ldc + col0]      = (CT)x0;
    C[(size_t)row*ldc + col0 + 16] = (CT)x1;
  }
}

// ----------------------------------------------------------------- the scan
// 8 blocks (1 per head, 1 per WGP). glu_W[h] (256KB bf16) resident in LDS in
// WMMA B-fragment layout. Normalized activations kept as a 16x256 bf16 tile
// (rows 4..15 pre-zeroed) so A fragments are two ds_load_b128, branch-free.
// LayerNorms use single-pass sum/sumsq reductions (2 barriers per LN).
__global__ __launch_bounds__(256) void k_scan(
    const float*  __restrict__ tokens,  // [NTOK][HD]
    const __bf16* __restrict__ gluW,    // [H][D][2D] bf16
    const float*  __restrict__ glub,    // [H][2D]
    const float*  __restrict__ sc, const float* __restrict__ ii,
    const float*  __restrict__ lng, const float* __restrict__ lnb,
    const float*  __restrict__ lsg, const float* __restrict__ lsb,
    float* __restrict__ states) {       // [NTOK][HD]
  extern __shared__ __align__(32) char smem[];
  __bf16* Bfrag = (__bf16*)smem;                     // 256 frags * 512 bf16 = 256KB
  __bf16* nrmb  = (__bf16*)(smem + 262144);          // [16][256] bf16, rows 4..15 zero
  float*  proj  = (float*)(smem + 262144 + 8192);    // [4][512]
  float*  red   = proj + 4*D2;                       // [256] partial sums
  float*  red2  = red + 256;                         // [256] partial sumsq
  float*  stats = red2 + 256;                        // [16]

  const int h = blockIdx.x, tid = threadIdx.x;
  const int wave = tid >> 5, lane = tid & 31;
  const int b = tid >> 6, d0 = (tid & 63) * 4;
  const int r15 = lane & 15, half = lane >> 4;

  // preload + swizzle glu weights into fragment layout
  const __bf16* g = gluW + (size_t)h * D_ * D2;
  for (int idx = tid; idx < D_*D2; idx += 256) {
    int k = idx / D2, n = idx - k*D2;
    int kt = k >> 5, kkk = k & 31, nt = n >> 4, nn = n & 15;
    int frag = kt*32 + nt;
    int fl = ((kkk >= 16) ? 16 : 0) + nn;
    Bfrag[(size_t)frag*512 + fl*16 + (kkk & 15)] = g[idx];
  }
  // zero pad rows 4..15 of the A tile (once)
  for (int idx = tid; idx < 12*D_; idx += 256) nrmb[4*D_ + idx] = f2bf(0.f);

  float st[4] = {0.f, 0.f, 0.f, 0.f};
  float scv[4], iiv[4], lgg[4], lgb[4], lsgv[4], lsbv[4], gba[4], gbb[4];
  #pragma unroll
  for (int j = 0; j < 4; ++j) {
    int d = d0 + j;
    scv[j] = sc[h*D_+d]; iiv[j] = ii[h*D_+d];
    lgg[j] = lng[d]; lgb[j] = lnb[d];
    lsgv[j] = lsg[d]; lsbv[j] = lsb[d];
    gba[j] = glub[h*D2 + d]; gbb[j] = glub[h*D2 + D_ + d];
  }
  __syncthreads();

  for (int t = 0; t < S_; ++t) {
    // ns = state*sc + x_t*ii, then LN(ln_glu) -> bf16 A tile
    float ns[4]; float psum = 0.f, psq = 0.f;
    const float* xrow = tokens + (size_t)(b*S_ + t)*HD + h*D_;
    #pragma unroll
    for (int j = 0; j < 4; ++j) {
      ns[j] = st[j]*scv[j] + xrow[d0+j]*iiv[j];
      psum += ns[j]; psq += ns[j]*ns[j];
    }
    red[tid] = psum; red2[tid] = psq; __syncthreads();
    if (tid < 4) {
      float s = 0.f, q = 0.f;
      for (int i = 0; i < 64; ++i) { s += red[tid*64+i]; q += red2[tid*64+i]; }
      float m = s*(1.f/256.f);
      stats[tid] = m;
      stats[8+tid] = rsqrtf(fmaxf(q*(1.f/256.f) - m*m, 0.f) + EPS_);
    }
    __syncthreads();
    float mean = stats[b], rinv = stats[8+b];
    {
      v4bf nv;
      #pragma unroll
      for (int j = 0; j < 4; ++j) nv[j] = f2bf((ns[j]-mean)*rinv*lgg[j] + lgb[j]);
      *(v4bf*)&nrmb[b*D_ + d0] = nv;   // one ds_store_b64
    }
    __syncthreads();

    // proj[16x512] = nrm(16x256, rows 4..15 zero) @ gluW_h(256x512)
    v8f acc[4] = {{},{},{},{}};
    #pragma unroll 1
    for (int kt = 0; kt < 8; ++kt) {
      const __bf16* ap = &nrmb[r15*D_ + kt*32 + half*16];
      v16bf a = cat16(*(const v8bf*)ap, *(const v8bf*)(ap + 8));
      #pragma unroll
      for (int i = 0; i < 4; ++i) {
        int nt = wave*4 + i;
        v16bf bb = *(const v16bf*)&Bfrag[(size_t)(kt*32 + nt)*512 + lane*16];
        acc[i] = __builtin_amdgcn_wmma_f32_16x16x32_bf16(false, a, false, bb, (short)0, acc[i], false, false);
      }
    }
    if (half == 0) {
      #pragma unroll
      for (int i = 0; i < 4; ++i) {
        int nc = (wave*4 + i)*16 + r15;
        #pragma unroll
        for (int v = 0; v < 4; ++v) proj[v*D2 + nc] = acc[i][v];
      }
    }
    __syncthreads();

    // silu(xa)*xb, LN(ln_state) -> out/state
    float gv[4]; psum = 0.f; psq = 0.f;
    #pragma unroll
    for (int j = 0; j < 4; ++j) {
      float xa = proj[b*D2 + d0 + j] + gba[j];
      float xb = proj[b*D2 + D_ + d0 + j] + gbb[j];
      gv[j] = xa * fast_sigmoid(xa) * xb;
      psum += gv[j]; psq += gv[j]*gv[j];
    }
    red[tid] = psum; red2[tid] = psq; __syncthreads();
    if (tid < 4) {
      float s = 0.f, q = 0.f;
      for (int i = 0; i < 64; ++i) { s += red[tid*64+i]; q += red2[tid*64+i]; }
      float m = s*(1.f/256.f);
      stats[tid] = m;
      stats[8+tid] = rsqrtf(fmaxf(q*(1.f/256.f) - m*m, 0.f) + EPS_);
    }
    __syncthreads();
    mean = stats[b]; rinv = stats[8+b];
    float* orow = states + (size_t)(b*S_ + t)*HD + h*D_;
    #pragma unroll
    for (int j = 0; j < 4; ++j) {
      float o = (gv[j]-mean)*rinv*lsgv[j] + lsbv[j];
      st[j] = o;
      orow[d0+j] = o;
    }
    __syncthreads();
  }
}

// ----------------------------------------- states*osh -> per-head LN -> bf16
__global__ __launch_bounds__(256) void k_preff(
    const float* __restrict__ states, const float* __restrict__ osh,
    const float* __restrict__ fg, const float* __restrict__ fb,
    __bf16* __restrict__ anorm) {
  __shared__ float red[256];
  __shared__ float red2[256];
  int token = blockIdx.x >> 3, h = blockIdx.x & 7, d = threadIdx.x;
  size_t idx = (size_t)token*HD + h*D_ + d;
  float v = states[idx] * osh[h*D_+d];
  red[d] = v; red2[d] = v*v; __syncthreads();
  for (int s = 128; s > 0; s >>= 1) {
    if (d < s) { red[d] += red[d+s]; red2[d] += red2[d+s]; }
    __syncthreads();
  }
  float mean = red[0]*(1.f/256.f);
  float rinv = rsqrtf(fmaxf(red2[0]*(1.f/256.f) - mean*mean, 0.f) + EPS_);
  anorm[idx] = f2bf((v-mean)*rinv*fg[h*D_+d] + fb[h*D_+d]);
}

// ------------------------------------ logits -> softmax(H) -> scale hid(bf16)
__global__ __launch_bounds__(256) void k_softmax(
    const __bf16* __restrict__ hid, const float* __restrict__ vvec,
    const float* __restrict__ cvec, __bf16* __restrict__ hid2,
    float* __restrict__ scores) {
  __shared__ float red[256];
  __shared__ float sw[8];
  int token = blockIdx.x, tid = threadIdx.x;
  int h = tid >> 5, j = tid & 31;
  const __bf16* row = hid + (size_t)token*HD;
  float p = 0.f;
  #pragma unroll
  for (int q = 0; q < 8; ++q) { int d = j*8 + q; p += bf2f(row[h*D_+d]) * vvec[h*D_+d]; }
  red[tid] = p; __syncthreads();
  if (tid < 8) { float s=0.f; for (int i=0;i<32;++i) s += red[tid*32+i]; sw[tid] = s + cvec[tid]; }
  __syncthreads();
  if (tid == 0) {
    float m = sw[0]; for (int i=1;i<8;++i) m = fmaxf(m, sw[i]);
    float z = 0.f, ev[8];
    for (int i=0;i<8;++i){ ev[i] = __expf(sw[i]-m); z += ev[i]; }
    float inv = __builtin_amdgcn_rcpf(z);
    for (int i=0;i<8;++i) sw[i] = ev[i]*inv;
  }
  __syncthreads();
  if (tid < 8) scores[(size_t)token*8 + tid] = sw[tid];
  #pragma unroll
  for (int q = 0; q < 8; ++q) {
    int col = tid*8 + q;
    hid2[(size_t)token*HD + col] = f2bf(bf2f(row[col]) * sw[col >> 8]);
  }
}

// -------------------------------------- + scores@ff_b2, final LN over E -> out
__global__ __launch_bounds__(256) void k_finalln(
    const float* __restrict__ weighted, const float* __restrict__ scores,
    const float* __restrict__ ffb2, const float* __restrict__ gg,
    const float* __restrict__ bb, float* __restrict__ out) {
  __shared__ float red[256];
  __shared__ float red2[256];
  __shared__ float sv[8];
  int token = blockIdx.x, tid = threadIdx.x;
  if (tid < 8) sv[tid] = scores[(size_t)token*8 + tid];
  __syncthreads();
  float pre[4]; float psum = 0.f, psq = 0.f;
  #pragma unroll
  for (int q = 0; q < 4; ++q) {
    int e = tid*4 + q;
    float p = weighted[(size_t)token*E_ + e];
    #pragma unroll
    for (int hh = 0; hh < 8; ++hh) p += sv[hh] * ffb2[(size_t)hh*E_ + e];
    pre[q] = p; psum += p; psq += p*p;
  }
  red[tid] = psum; red2[tid] = psq; __syncthreads();
  for (int s = 128; s > 0; s >>= 1) {
    if (tid < s) { red[tid] += red[tid+s]; red2[tid] += red2[tid+s]; }
    __syncthreads();
  }
  float mean = red[0]*(1.f/1024.f);
  float rinv = rsqrtf(fmaxf(red2[0]*(1.f/1024.f) - mean*mean, 0.f) + EPS_);
  #pragma unroll
  for (int q = 0; q < 4; ++q) {
    int e = tid*4 + q;
    out[(size_t)token*E_ + e] = (pre[q]-mean)*rinv*gg[e] + bb[e];
  }
}

// ---------------------------------------------------------------------------
extern "C" void kernel_launch(void* const* d_in, const int* in_sizes, int n_in,
                              void* d_out, int out_size, void* d_ws, size_t ws_size,
                              hipStream_t stream) {
  const float* emb    = (const float*)d_in[0];
  const float* sc_in  = (const float*)d_in[1];
  const float* ii_in  = (const float*)d_in[2];
  const float* osh_in = (const float*)d_in[3];
  const float* We2s   = (const float*)d_in[4];
  const float* be2s   = (const float*)d_in[5];
  const float* gluW   = (const float*)d_in[6];
  const float* glub   = (const float*)d_in[7];
  const float* lngg   = (const float*)d_in[8];
  const float* lngb   = (const float*)d_in[9];
  const float* lnsg   = (const float*)d_in[10];
  const float* lnsb   = (const float*)d_in[11];
  const float* fflng  = (const float*)d_in[12];
  const float* fflnb  = (const float*)d_in[13];
  const float* ffW1   = (const float*)d_in[14];
  const float* ffb1   = (const float*)d_in[15];
  const float* ffW2   = (const float*)d_in[16];
  const float* ffb2   = (const float*)d_in[17];
  const float* attnw  = (const float*)d_in[18];
  const float* attnb  = (const float*)d_in[19];
  const float* outg   = (const float*)d_in[20];
  const float* outb   = (const float*)d_in[21];

  char* w = (char*)d_ws; size_t off = 0;
  auto take = [&](size_t bytes) -> void* {
    void* p = w + off;
    off = (off + bytes + 255) & ~(size_t)255;
    return p;
  };
  __bf16* X_bf    = (__bf16*)take((size_t)NTOK*E_*2);
  __bf16* We2s_bf = (__bf16*)take((size_t)E_*HD*2);
  __bf16* gluW_bf = (__bf16*)take((size_t)H_*D_*D2*2);
  __bf16* ffW1_bf = (__bf16*)take((size_t)H_*D_*D_*2);
  __bf16* ffW2_bf = (__bf16*)take((size_t)H_*D_*E_*2);
  float*  sc      = (float*)take((size_t)HD*4);
  float*  ii      = (float*)take((size_t)HD*4);
  float*  osh     = (float*)take((size_t)HD*4);
  float*  vvec    = (float*)take((size_t)HD*4);
  float*  cvec    = (float*)take(64);
  float*  tokens  = (float*)take((size_t)NTOK*HD*4);
  float*  states  = (float*)take((size_t)NTOK*HD*4);
  __bf16* anorm   = (__bf16*)take((size_t)NTOK*HD*2);
  __bf16* hid_bf  = (__bf16*)take((size_t)NTOK*HD*2);
  __bf16* hid2_bf = (__bf16*)take((size_t)NTOK*HD*2);
  float*  scores  = (float*)take((size_t)NTOK*H_*4);
  float*  weighted= (float*)take((size_t)NTOK*E_*4);

  // bf16 conversions
  k_f32_to_bf16<<<1024, 256, 0, stream>>>(emb,  X_bf,    NTOK*E_);
  k_f32_to_bf16<<<1024, 256, 0, stream>>>(We2s, We2s_bf, E_*HD);
  k_f32_to_bf16<<<1024, 256, 0, stream>>>(gluW, gluW_bf, H_*D_*D2);
  k_f32_to_bf16<<<1024, 256, 0, stream>>>(ffW1, ffW1_bf, H_*D_*D_);
  k_f32_to_bf16<<<1024, 256, 0, stream>>>(ffW2, ffW2_bf, H_*D_*E_);

  // gate row-sums + logit projection vector
  k_prep<<<H_, 256, 0, stream>>>(sc_in, ii_in, osh_in, ffW2, ffb2, attnw, attnb,
                                 sc, ii, osh, vvec, cvec);

  // tokens[4096,2048] = X @ W_e2s + b
  k_gemm<0, float><<<dim3(NTOK/64, HD/64), 256, 0, stream>>>(
      X_bf, E_, We2s_bf, HD, tokens, HD, be2s, E_);

  // sequential scan, one head per WGP, glu weights resident in LDS
  size_t scan_smem = 262144 + 8192 /*nrmb*/ + (4*D2 + 256 + 256 + 16)*4;
  hipFuncSetAttribute((const void*)k_scan,
                      hipFuncAttributeMaxDynamicSharedMemorySize, (int)scan_smem);
  k_scan<<<H_, 256, scan_smem, stream>>>(tokens, gluW_bf, glub, sc, ii,
                                         lngg, lngb, lnsg, lnsb, states);

  // states*osh -> per-head LN -> bf16
  k_preff<<<NTOK*H_, 256, 0, stream>>>(states, osh, fflng, fflnb, anorm);

  // per-head hid = gelu(anorm @ ff_W1 + b1), bf16 out
  for (int h = 0; h < H_; ++h) {
    k_gemm<1, __bf16><<<dim3(NTOK/64, D_/64), 256, 0, stream>>>(
        anorm + h*D_, HD, ffW1_bf + (size_t)h*D_*D_, D_,
        hid_bf + h*D_, HD, ffb1 + h*D_, D_);
  }

  // logits -> softmax over heads -> hid2 = scores * hid
  k_softmax<<<NTOK, 256, 0, stream>>>(hid_bf, vvec, cvec, hid2_bf, scores);

  // weighted[4096,1024] = hid2[4096,2048] @ ff_W2_flat[2048,1024]
  k_gemm<2, float><<<dim3(NTOK/64, E_/64), 256, 0, stream>>>(
      hid2_bf, HD, ffW2_bf, E_, weighted, E_, nullptr, HD);

  // + scores@ff_b2, final LN -> out
  k_finalln<<<NTOK, 256, 0, stream>>>(weighted, scores, ffb2, outg, outb,
                                      (float*)d_out);
}
